// GNNBackbone_24824910971340
// MI455X (gfx1250) — matrix operand order repaired and checked by reading, not compiled, not run on hardware
//
#include <hip/hip_runtime.h>
#include <hip/hip_bf16.h>
#include <math.h>

typedef __attribute__((ext_vector_type(16))) __bf16 v16bf;
typedef __attribute__((ext_vector_type(8)))  __bf16 v8bf;
typedef __attribute__((ext_vector_type(8)))  float  v8f;

constexpr int kD   = 128;        // feature dim
constexpr int kV   = 128;        // vars per graph
constexpr int kB   = 512;        // graphs
constexpr int kN   = kB * kV;    // 65536 nodes
constexpr int kE   = 2 * kN;     // 131072 edges
constexpr int kEPG = kE / kB;    // 256 edges per graph
constexpr int kH   = 4;          // heads
constexpr float kEps = 1e-5f;

// ---------------------------------------------------------------------------
// WMMA fragment loaders (CDNA5 16-bit layouts, wave32)
// A 16x32: lanes 0-15 -> K {0..7, 16..23}; lanes 16-31 -> K {8..15, 24..31}
// B 32x16: lanes 0-15 -> K {0..7, 16..23}; lanes 16-31 -> K {8..15, 24..31}
// ---------------------------------------------------------------------------
static __device__ __forceinline__ v16bf load_a_frag(const __bf16* base, int ld,
                                                    int lane, int k0) {
  int r  = lane & 15;
  int kb = (lane >> 4) << 3;  // 0 or 8
  const __bf16* p = base + (size_t)r * ld + (k0 + kb);
  v8bf lo = *(const v8bf*)(p);
  v8bf hi = *(const v8bf*)(p + 16);
  v16bf f;
#pragma unroll
  for (int i = 0; i < 8; ++i) { f[i] = lo[i]; f[i + 8] = hi[i]; }
  return f;
}

static __device__ __forceinline__ v16bf load_b_frag(const __bf16* base, int ld,
                                                    int lane, int k0) {
  int n  = lane & 15;
  int kb = (lane >> 4) << 3;  // 0 or 8
  const __bf16* p = base + (size_t)n * ld + (k0 + kb);
  v8bf lo = *(const v8bf*)(p);
  v8bf hi = *(const v8bf*)(p + 16);
  v16bf f;
#pragma unroll
  for (int i = 0; i < 8; ++i) { f[i] = lo[i]; f[i + 8] = hi[i]; }
  return f;
}

static __device__ __forceinline__ v8f wmma_bf16(v16bf a, v16bf b, v8f c) {
  return __builtin_amdgcn_wmma_f32_16x16x32_bf16(false, a, false, b, (short)0,
                                                 c, false, false);
}

static __device__ __forceinline__ float wave_allsum(float v) {
#pragma unroll
  for (int m = 16; m >= 1; m >>= 1) v += __shfl_xor(v, m, 32);
  return v;
}

// ---------------------------------------------------------------------------
// Generic bf16 GEMM: C[M,N] = act(A[M,K] @ Wt[N,K]^T + bias + extra_row)
// OUT: 0 -> f32 row-major (ldc = row stride)
//      1 -> bf16 row-major (ldc = row stride)
//      2 -> bf16 transposed: Ct[col*ldc + row] (ldc = M total)
// grid = (N/32, M/64), block = 256 (8 waves, 4x2 tile grid of 16x16 tiles)
// ---------------------------------------------------------------------------
template <int OUT>
__global__ __launch_bounds__(256) void gemm_bf16(
    const __bf16* __restrict__ A, int lda, const __bf16* __restrict__ Bt,
    int ldb, float* __restrict__ Cf, __bf16* __restrict__ Cb, size_t ldc,
    int K, const float* __restrict__ bias, const float* __restrict__ extra,
    int relu) {
  int lane = threadIdx.x & 31;
  int wid  = threadIdx.x >> 5;
  int m0 = ((int)blockIdx.y * 4 + (wid >> 1)) * 16;
  int n0 = ((int)blockIdx.x * 2 + (wid & 1)) * 16;
  const __bf16* Ab = A + (size_t)m0 * lda;
  const __bf16* Bb = Bt + (size_t)n0 * ldb;
  v8f acc;
#pragma unroll
  for (int i = 0; i < 8; ++i) acc[i] = 0.0f;
  for (int k0 = 0; k0 < K; k0 += 32) {
    v16bf af = load_a_frag(Ab, lda, lane, k0);
    v16bf bfr = load_b_frag(Bb, ldb, lane, k0);
    acc = wmma_bf16(af, bfr, acc);
  }
  int col = n0 + (lane & 15);
  float bv = bias ? bias[col] : 0.0f;
  float ev = extra ? extra[col] : 0.0f;
  int rb = m0 + ((lane >> 4) << 3);
#pragma unroll
  for (int r = 0; r < 8; ++r) {
    float v = acc[r] + bv + ev;
    if (relu) v = fmaxf(v, 0.0f);
    size_t row = (size_t)(rb + r);
    if (OUT == 0)      Cf[row * ldc + col] = v;
    else if (OUT == 1) Cb[row * ldc + col] = (__bf16)v;
    else               Cb[(size_t)col * ldc + row] = (__bf16)v;
  }
}

// ---------------------------------------------------------------------------
// Fold broadcast-constant concat blocks into per-MLP constant rows.
// econst = eb1 + edge_embed@eW1[0:D] + global@eW1[3D:4D]
// nconst = nb1 + global@nW1[3D:4D];  gconst = gb1 + global@gW1[2D:3D]
// ---------------------------------------------------------------------------
__global__ void consts_kernel(const float* __restrict__ ee,
                              const float* __restrict__ ge,
                              const float* __restrict__ eW1,
                              const float* __restrict__ eb1,
                              const float* __restrict__ nW1,
                              const float* __restrict__ nb1,
                              const float* __restrict__ gW1,
                              const float* __restrict__ gb1, float* econst,
                              float* nconst, float* gconst) {
  int d = threadIdx.x;
  float e = eb1[d], n = nb1[d], g = gb1[d];
  for (int k = 0; k < kD; ++k) {
    float gv = ge[k];
    e += ee[k] * eW1[k * kD + d] + gv * eW1[(3 * kD + k) * kD + d];
    n += gv * nW1[(3 * kD + k) * kD + d];
    g += gv * gW1[(2 * kD + k) * kD + d];
  }
  econst[d] = e; nconst[d] = n; gconst[d] = g;
}

// Wt[n*K + k] = (bf16) W[k*stride + n]  (WMMA-friendly transposed weights)
__global__ void packWt_kernel(const float* __restrict__ W,
                              __bf16* __restrict__ Wt, int K, int stride,
                              int total) {
  int t = blockIdx.x * blockDim.x + threadIdx.x;
  if (t >= total) return;
  int n = t / K, k = t - n * K;
  Wt[t] = (__bf16)W[(size_t)k * stride + n];
}

__global__ __launch_bounds__(256) void gather_nodes_kernel(
    const int* __restrict__ ids, const float* __restrict__ embed,
    float* __restrict__ nodes_f, __bf16* __restrict__ nodes_b) {
  size_t idx = (size_t)blockIdx.x * blockDim.x + threadIdx.x;  // N*D
  size_t n = idx >> 7; int d = (int)(idx & 127);
  float v = embed[(size_t)ids[n] * kD + d];
  nodes_f[idx] = v;
  nodes_b[idx] = (__bf16)v;
}

__global__ __launch_bounds__(256) void build_acat_e_kernel(
    const int* __restrict__ snd, const int* __restrict__ rcv,
    const __bf16* __restrict__ nodes_b, __bf16* __restrict__ acat) {
  size_t idx = (size_t)blockIdx.x * blockDim.x + threadIdx.x;  // E*D
  size_t e = idx >> 7; int d = (int)(idx & 127);
  acat[e * 256 + d]       = nodes_b[(size_t)snd[e] * kD + d];
  acat[e * 256 + 128 + d] = nodes_b[(size_t)rcv[e] * kD + d];
}

__global__ __launch_bounds__(256) void scatter_edges_kernel(
    const int* __restrict__ snd, const int* __restrict__ rcv,
    const float* __restrict__ eu, float* __restrict__ sent,
    float* __restrict__ recv, float* __restrict__ eagg) {
  size_t idx = (size_t)blockIdx.x * blockDim.x + threadIdx.x;  // E*D
  size_t e = idx >> 7; int d = (int)(idx & 127);
  float v = eu[idx];
  atomicAdd(&sent[(size_t)snd[e] * kD + d], v);
  atomicAdd(&recv[(size_t)rcv[e] * kD + d], v);
  atomicAdd(&eagg[(e / kEPG) * kD + d], v);
}

__global__ __launch_bounds__(256) void build_acat_n_kernel(
    const __bf16* __restrict__ nodes_b, const float* __restrict__ sent,
    const float* __restrict__ recv, __bf16* __restrict__ acat) {
  size_t idx = (size_t)blockIdx.x * blockDim.x + threadIdx.x;  // N*D
  size_t n = idx >> 7; int d = (int)(idx & 127);
  acat[n * 384 + d]       = nodes_b[idx];
  acat[n * 384 + 128 + d] = (__bf16)sent[idx];
  acat[n * 384 + 256 + d] = (__bf16)recv[idx];
}

__global__ __launch_bounds__(256) void nagg_scatter_kernel(
    const float* __restrict__ nu, float* __restrict__ nagg) {
  size_t idx = (size_t)blockIdx.x * blockDim.x + threadIdx.x;  // N*D
  size_t n = idx >> 7; int d = (int)(idx & 127);
  atomicAdd(&nagg[(n / kV) * kD + d], nu[idx]);
}

__global__ __launch_bounds__(256) void build_acat_g_kernel(
    const float* __restrict__ nagg, const float* __restrict__ eagg,
    __bf16* __restrict__ acat) {
  size_t idx = (size_t)blockIdx.x * blockDim.x + threadIdx.x;  // B*D
  size_t b = idx >> 7; int d = (int)(idx & 127);
  acat[b * 256 + d]       = (__bf16)nagg[idx];
  acat[b * 256 + 128 + d] = (__bf16)eagg[idx];
}

// residual + LayerNorm over D=128, one wave per row (4 elems / lane, wave32)
__global__ __launch_bounds__(256) void ln_nodes_kernel(
    const float* __restrict__ X, const float* __restrict__ U,
    const float* __restrict__ s, const float* __restrict__ o,
    __bf16* __restrict__ nf) {
  int lane = threadIdx.x & 31, wid = threadIdx.x >> 5;
  size_t row = (size_t)blockIdx.x * 8 + wid;
  int d0 = lane * 4;
  float x[4], sm = 0.f, sq = 0.f;
#pragma unroll
  for (int i = 0; i < 4; ++i) {
    x[i] = X[row * kD + d0 + i] + U[row * kD + d0 + i];
    sm += x[i]; sq += x[i] * x[i];
  }
  sm = wave_allsum(sm); sq = wave_allsum(sq);
  float mean = sm * (1.f / kD);
  float inv = rsqrtf(sq * (1.f / kD) - mean * mean + kEps);
#pragma unroll
  for (int i = 0; i < 4; ++i)
    nf[row * kD + d0 + i] =
        (__bf16)((x[i] - mean) * inv * s[d0 + i] + o[d0 + i]);
}

__global__ __launch_bounds__(256) void ln_out_kernel(
    const float* __restrict__ X, const float* __restrict__ s,
    const float* __restrict__ o, float* __restrict__ out) {
  int lane = threadIdx.x & 31, wid = threadIdx.x >> 5;
  size_t row = (size_t)blockIdx.x * 8 + wid;
  int d0 = lane * 4;
  float x[4], sm = 0.f, sq = 0.f;
#pragma unroll
  for (int i = 0; i < 4; ++i) {
    x[i] = X[row * kD + d0 + i];
    sm += x[i]; sq += x[i] * x[i];
  }
  sm = wave_allsum(sm); sq = wave_allsum(sq);
  float mean = sm * (1.f / kD);
  float inv = rsqrtf(sq * (1.f / kD) - mean * mean + kEps);
#pragma unroll
  for (int i = 0; i < 4; ++i)
    out[row * kD + d0 + i] = (x[i] - mean) * inv * s[d0 + i] + o[d0 + i];
}

// glob residual + LN(ln_g) then LN(ln_og) -> second output
__global__ __launch_bounds__(256) void ln_glob_kernel(
    const float* __restrict__ ge, const float* __restrict__ gu,
    const float* __restrict__ s1, const float* __restrict__ o1,
    const float* __restrict__ s2, const float* __restrict__ o2,
    float* __restrict__ out) {
  int lane = threadIdx.x & 31, wid = threadIdx.x >> 5;
  size_t row = (size_t)blockIdx.x * 8 + wid;
  int d0 = lane * 4;
  float x[4], sm = 0.f, sq = 0.f;
#pragma unroll
  for (int i = 0; i < 4; ++i) {
    x[i] = ge[d0 + i] + gu[row * kD + d0 + i];
    sm += x[i]; sq += x[i] * x[i];
  }
  sm = wave_allsum(sm); sq = wave_allsum(sq);
  float mean = sm * (1.f / kD);
  float inv = rsqrtf(sq * (1.f / kD) - mean * mean + kEps);
  float y[4]; sm = 0.f; sq = 0.f;
#pragma unroll
  for (int i = 0; i < 4; ++i) {
    y[i] = (x[i] - mean) * inv * s1[d0 + i] + o1[d0 + i];
    sm += y[i]; sq += y[i] * y[i];
  }
  sm = wave_allsum(sm); sq = wave_allsum(sq);
  mean = sm * (1.f / kD);
  inv = rsqrtf(sq * (1.f / kD) - mean * mean + kEps);
#pragma unroll
  for (int i = 0; i < 4; ++i)
    out[row * kD + d0 + i] = (y[i] - mean) * inv * s2[d0 + i] + o2[d0 + i];
}

// ---------------------------------------------------------------------------
// Attention per (head, graph): logits = qh@kh^T * 1/sqrt(32) -> softmax ->
// attn@vh. Probabilities staged in LDS (bf16). vht is V transposed [D x N].
// grid = (H, B), block = 256 (8 waves, wave w owns rows [w*16, w*16+16)).
// ---------------------------------------------------------------------------
__global__ __launch_bounds__(256) void attention_kernel(
    const __bf16* __restrict__ qh, const __bf16* __restrict__ kh,
    const __bf16* __restrict__ vht, __bf16* __restrict__ attn_out) {
  __shared__ __bf16 Sb[128][128];  // 32 KB
  int lane = threadIdx.x & 31, wid = threadIdx.x >> 5;
  int h = blockIdx.x, b = blockIdx.y;
  int m0 = wid * 16;
  const __bf16* Qb = qh + ((size_t)b * kV) * kD + (size_t)h * 32;
  const __bf16* Kb = kh + ((size_t)b * kV) * kD + (size_t)h * 32;

  // phase 1: logits (M=128,N=128,K=32), one WMMA per 16x16 tile
  v16bf af = load_a_frag(Qb + (size_t)m0 * kD, kD, lane, 0);
  v8f accs[8];
#pragma unroll
  for (int tn = 0; tn < 8; ++tn) {
    v16bf bfr = load_b_frag(Kb + (size_t)(tn * 16) * kD, kD, lane, 0);
    v8f z;
#pragma unroll
    for (int i = 0; i < 8; ++i) z[i] = 0.0f;
    accs[tn] = wmma_bf16(af, bfr, z);
  }
  const float scale = 0.17677669529663687f;  // 1/sqrt(32)
  int rb = m0 + ((lane >> 4) << 3);
#pragma unroll
  for (int r = 0; r < 8; ++r) {
    float mx = -3.4e38f;
#pragma unroll
    for (int tn = 0; tn < 8; ++tn) mx = fmaxf(mx, accs[tn][r] * scale);
#pragma unroll
    for (int m = 1; m < 16; m <<= 1) mx = fmaxf(mx, __shfl_xor(mx, m, 32));
    float e[8], sum = 0.f;
#pragma unroll
    for (int tn = 0; tn < 8; ++tn) {
      e[tn] = __expf(accs[tn][r] * scale - mx);
      sum += e[tn];
    }
#pragma unroll
    for (int m = 1; m < 16; m <<= 1) sum += __shfl_xor(sum, m, 32);
    float isum = 1.0f / sum;
    int row = rb + r;
#pragma unroll
    for (int tn = 0; tn < 8; ++tn)
      Sb[row][tn * 16 + (lane & 15)] = (__bf16)(e[tn] * isum);
  }
  __syncthreads();

  // phase 2: out = P @ V  (M=128, N=32, K=128), A from LDS, B from vht
  const __bf16* Sbase = &Sb[0][0] + (size_t)m0 * 128;
  const __bf16* Vb0 = vht + (size_t)(h * 32) * kN + (size_t)b * kV;
  const __bf16* Vb1 = Vb0 + (size_t)16 * kN;
  v8f o0, o1;
#pragma unroll
  for (int i = 0; i < 8; ++i) { o0[i] = 0.0f; o1[i] = 0.0f; }
#pragma unroll
  for (int k0 = 0; k0 < 128; k0 += 32) {
    v16bf a2 = load_a_frag(Sbase, 128, lane, k0);
    v16bf b0 = load_b_frag(Vb0, kN, lane, k0);
    v16bf b1 = load_b_frag(Vb1, kN, lane, k0);
    o0 = wmma_bf16(a2, b0, o0);
    o1 = wmma_bf16(a2, b1, o1);
  }
#pragma unroll
  for (int r = 0; r < 8; ++r) {
    size_t t = (size_t)(rb + r);
    size_t base = ((size_t)b * kV + t) * kD + (size_t)h * 32;
    attn_out[base + (lane & 15)]      = (__bf16)o0[r];
    attn_out[base + 16 + (lane & 15)] = (__bf16)o1[r];
  }
}

// ---------------------------------------------------------------------------
extern "C" void kernel_launch(void* const* d_in, const int* in_sizes, int n_in,
                              void* d_out, int out_size, void* d_ws,
                              size_t ws_size, hipStream_t stream) {
  (void)in_sizes; (void)n_in; (void)out_size; (void)ws_size;
  const int*   node_ids = (const int*)d_in[0];
  const int*   senders  = (const int*)d_in[1];
  const int*   receivers= (const int*)d_in[2];
  const float* embed    = (const float*)d_in[4];
  const float* edge_emb = (const float*)d_in[5];
  const float* glob_emb = (const float*)d_in[6];
  const float* eW1 = (const float*)d_in[7];
  const float* eb1 = (const float*)d_in[8];
  const float* eW2 = (const float*)d_in[9];
  const float* eb2 = (const float*)d_in[10];
  const float* nW1 = (const float*)d_in[11];
  const float* nb1 = (const float*)d_in[12];
  const float* nW2 = (const float*)d_in[13];
  const float* nb2 = (const float*)d_in[14];
  const float* gW1 = (const float*)d_in[15];
  const float* gb1 = (const float*)d_in[16];
  const float* gW2 = (const float*)d_in[17];
  const float* gb2 = (const float*)d_in[18];
  const float* ln_n_s = (const float*)d_in[19];
  const float* ln_n_o = (const float*)d_in[20];
  const float* ln_g_s = (const float*)d_in[21];
  const float* ln_g_o = (const float*)d_in[22];
  const float* pW = (const float*)d_in[23];
  const float* pb = (const float*)d_in[24];
  const float* Wq = (const float*)d_in[25];
  const float* bq = (const float*)d_in[26];
  const float* Wk = (const float*)d_in[27];
  const float* bk = (const float*)d_in[28];
  const float* Wv = (const float*)d_in[29];
  const float* bv = (const float*)d_in[30];
  const float* Wo = (const float*)d_in[31];
  const float* bo = (const float*)d_in[32];
  const float* ln_on_s = (const float*)d_in[33];
  const float* ln_on_o = (const float*)d_in[34];
  const float* ln_og_s = (const float*)d_in[35];
  const float* ln_og_o = (const float*)d_in[36];
  float* out = (float*)d_out;

  // ---- workspace layout (regions reused over the pipeline's lifetime) ----
  const size_t MB = (size_t)1 << 20;
  char* W = (char*)d_ws;
  float*  nodes_f = (float*)(W);                 // [0,32MB)
  __bf16* nodes_b = (__bf16*)(W + 32 * MB);      // [32,48)
  __bf16* acat_e  = (__bf16*)(W + 48 * MB);      // [48,112) then acat_n
  __bf16* acat_n  = (__bf16*)(W + 48 * MB);
  float*  edges_u = (float*)(W + 112 * MB);      // [112,176) then qkv
  __bf16* qkv     = (__bf16*)(W + 112 * MB);
  __bf16* He      = (__bf16*)(W + 176 * MB);     // [176,208) then qh,kh
  __bf16* qh      = (__bf16*)(W + 176 * MB);
  __bf16* kh      = (__bf16*)(W + 192 * MB);
  float*  sent    = (float*)(W + 208 * MB);      // [208,240) then proj
  float*  proj    = (float*)(W + 208 * MB);
  float*  recv    = (float*)(W + 240 * MB);      // [240,272) then nf,vht
  __bf16* nf      = (__bf16*)(W + 240 * MB);
  __bf16* vht     = (__bf16*)(W + 256 * MB);
  __bf16* Hn      = (__bf16*)(W + 272 * MB);     // [272,288) then attn_out
  __bf16* attn_o  = (__bf16*)(W + 272 * MB);
  float*  nodes_u = (float*)(W + 288 * MB);      // [288,320)
  char* S = W + 320 * MB;
  auto alloc = [&](size_t bytes) -> char* {
    char* p = S;
    S += (bytes + 255) & ~(size_t)255;
    return p;
  };
  float*  eagg   = (float*)alloc((size_t)kB * kD * 4);
  float*  nagg   = (float*)alloc((size_t)kB * kD * 4);
  float*  glob_u = (float*)alloc((size_t)kB * kD * 4);
  __bf16* acat_g = (__bf16*)alloc((size_t)kB * 256 * 2);
  __bf16* Hg     = (__bf16*)alloc((size_t)kB * kD * 2);
  float*  econst = (float*)alloc(kD * 4);
  float*  nconst = (float*)alloc(kD * 4);
  float*  gconst = (float*)alloc(kD * 4);
  __bf16* Wt_e1 = (__bf16*)alloc(128 * 256 * 2);
  __bf16* Wt_e2 = (__bf16*)alloc(128 * 128 * 2);
  __bf16* Wt_n1 = (__bf16*)alloc(128 * 384 * 2);
  __bf16* Wt_n2 = (__bf16*)alloc(128 * 128 * 2);
  __bf16* Wt_g1 = (__bf16*)alloc(128 * 256 * 2);
  __bf16* Wt_g2 = (__bf16*)alloc(128 * 128 * 2);
  __bf16* Wt_p  = (__bf16*)alloc(384 * 128 * 2);
  __bf16* Wt_q  = (__bf16*)alloc(128 * 128 * 2);
  __bf16* Wt_k  = (__bf16*)alloc(128 * 128 * 2);
  __bf16* Wt_v  = (__bf16*)alloc(128 * 128 * 2);
  __bf16* Wt_o  = (__bf16*)alloc(128 * 128 * 2);

  // ---- zero the scatter accumulators ----
  hipMemsetAsync(sent, 0, (size_t)kN * kD * 4, stream);
  hipMemsetAsync(recv, 0, (size_t)kN * kD * 4, stream);
  hipMemsetAsync(eagg, 0, (size_t)kB * kD * 4, stream);
  hipMemsetAsync(nagg, 0, (size_t)kB * kD * 4, stream);

  // ---- constants + transposed bf16 weight packs ----
  consts_kernel<<<1, 128, 0, stream>>>(edge_emb, glob_emb, eW1, eb1, nW1, nb1,
                                       gW1, gb1, econst, nconst, gconst);
  auto packW = [&](const float* src, __bf16* dst, int K, int Nc, int stride) {
    int total = K * Nc;
    packWt_kernel<<<(total + 255) / 256, 256, 0, stream>>>(src, dst, K, stride,
                                                           total);
  };
  packW(eW1 + kD * kD, Wt_e1, 256, 128, kD);  // sender+receiver blocks
  packW(eW2, Wt_e2, 128, 128, kD);
  packW(nW1, Wt_n1, 384, 128, kD);            // node+sent+recv blocks
  packW(nW2, Wt_n2, 128, 128, kD);
  packW(gW1, Wt_g1, 256, 128, kD);            // nagg+eagg blocks
  packW(gW2, Wt_g2, 128, 128, kD);
  packW(pW, Wt_p, 128, 384, 384);
  packW(Wq, Wt_q, 128, 128, kD);
  packW(Wk, Wt_k, 128, 128, kD);
  packW(Wv, Wt_v, 128, 128, kD);
  packW(Wo, Wt_o, 128, 128, kD);

  // ---- embed gather, edge MLP ----
  gather_nodes_kernel<<<kN * kD / 256, 256, 0, stream>>>(node_ids, embed,
                                                         nodes_f, nodes_b);
  build_acat_e_kernel<<<kE * kD / 256, 256, 0, stream>>>(senders, receivers,
                                                         nodes_b, acat_e);
  gemm_bf16<1><<<dim3(4, kE / 64), 256, 0, stream>>>(
      acat_e, 256, Wt_e1, 256, nullptr, He, 128, 256, nullptr, econst, 1);
  gemm_bf16<0><<<dim3(4, kE / 64), 256, 0, stream>>>(
      He, 128, Wt_e2, 128, edges_u, nullptr, 128, 128, eb2, nullptr, 0);

  // ---- segment sums + node MLP ----
  scatter_edges_kernel<<<kE * kD / 256, 256, 0, stream>>>(senders, receivers,
                                                          edges_u, sent, recv,
                                                          eagg);
  build_acat_n_kernel<<<kN * kD / 256, 256, 0, stream>>>(nodes_b, sent, recv,
                                                         acat_n);
  gemm_bf16<1><<<dim3(4, kN / 64), 256, 0, stream>>>(
      acat_n, 384, Wt_n1, 384, nullptr, Hn, 128, 384, nullptr, nconst, 1);
  gemm_bf16<0><<<dim3(4, kN / 64), 256, 0, stream>>>(
      Hn, 128, Wt_n2, 128, nodes_u, nullptr, 128, 128, nb2, nullptr, 0);

  // ---- global MLP ----
  nagg_scatter_kernel<<<kN * kD / 256, 256, 0, stream>>>(nodes_u, nagg);
  build_acat_g_kernel<<<kB * kD / 256, 256, 0, stream>>>(nagg, eagg, acat_g);
  gemm_bf16<1><<<dim3(4, kB / 64), 256, 0, stream>>>(
      acat_g, 256, Wt_g1, 256, nullptr, Hg, 128, 256, nullptr, gconst, 1);
  gemm_bf16<0><<<dim3(4, kB / 64), 256, 0, stream>>>(
      Hg, 128, Wt_g2, 128, glob_u, nullptr, 128, 128, gb2, nullptr, 0);

  // ---- residual + LN ----
  ln_nodes_kernel<<<kN / 8, 256, 0, stream>>>(nodes_f, nodes_u, ln_n_s, ln_n_o,
                                              nf);
  ln_glob_kernel<<<kB / 8, 256, 0, stream>>>(glob_emb, glob_u, ln_g_s, ln_g_o,
                                             ln_og_s, ln_og_o,
                                             out + (size_t)kN * kD);

  // ---- projection + per-head QKV (V stored transposed for attention B) ----
  gemm_bf16<1><<<dim3(12, kN / 64), 256, 0, stream>>>(
      nf, 128, Wt_p, 128, nullptr, qkv, 384, 128, pb, nullptr, 0);
  gemm_bf16<1><<<dim3(4, kN / 64), 256, 0, stream>>>(
      qkv, 384, Wt_q, 128, nullptr, qh, 128, 128, bq, nullptr, 0);
  gemm_bf16<1><<<dim3(4, kN / 64), 256, 0, stream>>>(
      qkv + 128, 384, Wt_k, 128, nullptr, kh, 128, 128, bk, nullptr, 0);
  gemm_bf16<2><<<dim3(4, kN / 64), 256, 0, stream>>>(
      qkv + 256, 384, Wt_v, 128, nullptr, vht, (size_t)kN, 128, bv, nullptr, 0);

  // ---- attention + output projection + final LN ----
  attention_kernel<<<dim3(kH, kB), 256, 0, stream>>>(qh, kh, vht, attn_o);
  gemm_bf16<0><<<dim3(4, kN / 64), 256, 0, stream>>>(
      attn_o, 128, Wt_o, 128, proj, nullptr, 128, 128, bo, nullptr, 0);
  ln_out_kernel<<<kN / 8, 256, 0, stream>>>(proj, ln_on_s, ln_on_o, out);
}
